// GumbelSoftmaxVectorQuantizer_38001870635051
// MI455X (gfx1250) — compile-verified, matrix operand order checked by reading, not run on hardware
//
#include <hip/hip_runtime.h>
#include <hip/hip_bf16.h>

// ---- problem constants (from reference setup_inputs) ----
#define BT    16384      // B*T
#define DIN   1024       // K
#define GN    4
#define VN    1024
#define DN    128
#define NTOT  4096       // G*V (GEMM N)
#define INV_TAU 0.5f     // 1/TAU

typedef __attribute__((ext_vector_type(16))) __bf16 v16bf;
typedef __attribute__((ext_vector_type(8)))  __bf16 v8bf;
typedef __attribute__((ext_vector_type(8)))  float  v8f;

// ------------------------------------------------------------------
// Kernel 1: W (DIN x NTOT, f32, row-major) -> Wt (NTOT x DIN, bf16)
// 32x32 LDS tile transpose (33 stride to dodge bank conflicts).
// ------------------------------------------------------------------
__global__ __launch_bounds__(256)
void transpose_convert_w(const float* __restrict__ W, __bf16* __restrict__ Wt) {
    __shared__ float tile[32][33];
    const int tx = threadIdx.x & 31;
    const int ty = threadIdx.x >> 5;          // 0..7
    const int n0 = blockIdx.x * 32;           // N tile
    const int k0 = blockIdx.y * 32;           // K tile
#pragma unroll
    for (int i = 0; i < 32; i += 8)
        tile[ty + i][tx] = W[(size_t)(k0 + ty + i) * NTOT + (n0 + tx)];
    __syncthreads();
#pragma unroll
    for (int i = 0; i < 32; i += 8)
        Wt[(size_t)(n0 + ty + i) * DIN + (k0 + tx)] = (__bf16)tile[tx][ty + i];
}

// ------------------------------------------------------------------
// WMMA fragment helpers (A 16-bit 16x32 layout: per-lane K chunks at
// {base .. base+7} and {base+16 .. base+23}, base = (lane<16 ? 0 : 8);
// B mirrors A when sourced from Wt rows).
// ------------------------------------------------------------------
__device__ inline v16bf load_bf16_frag(const __bf16* __restrict__ p) {
    v8bf lo = *(const v8bf*)(p);
    v8bf hi = *(const v8bf*)(p + 16);
    v16bf r;
#pragma unroll
    for (int i = 0; i < 8; ++i) { r[i] = lo[i]; r[i + 8] = hi[i]; }
    return r;
}

__device__ inline v16bf load_f32_frag_cvt(const float* __restrict__ p) {
    const float4 f0 = *(const float4*)(p);
    const float4 f1 = *(const float4*)(p + 4);
    const float4 f2 = *(const float4*)(p + 16);
    const float4 f3 = *(const float4*)(p + 20);
    v16bf a;
    a[0]=(__bf16)f0.x;  a[1]=(__bf16)f0.y;  a[2]=(__bf16)f0.z;  a[3]=(__bf16)f0.w;
    a[4]=(__bf16)f1.x;  a[5]=(__bf16)f1.y;  a[6]=(__bf16)f1.z;  a[7]=(__bf16)f1.w;
    a[8]=(__bf16)f2.x;  a[9]=(__bf16)f2.y;  a[10]=(__bf16)f2.z; a[11]=(__bf16)f2.w;
    a[12]=(__bf16)f3.x; a[13]=(__bf16)f3.y; a[14]=(__bf16)f3.z; a[15]=(__bf16)f3.w;
    return a;
}

#define WMMA_BF16(ACC, AF, BF) \
    __builtin_amdgcn_wmma_f32_16x16x32_bf16(false, (AF), false, (BF), (short)0, (ACC), false, false)

// ------------------------------------------------------------------
// Kernel 2: logits = (A @ Wt^T + bias + gumbel) * INV_TAU
// A: (BT x DIN) f32 row-major; Wt: (NTOT x DIN) bf16 row-major.
// Block = 256 thr = 8 waves; wave tile = 32(M) x 64(N); block = 64x256.
// Two-stage software pipeline: prefetch k+32 fragments before the
// current step's 8 WMMAs. Grid = (NTOT/256, BT/64).
// ------------------------------------------------------------------
__global__ __launch_bounds__(256, 1)
void gemm_logits_wmma(const float* __restrict__ A,
                      const __bf16* __restrict__ Wt,
                      const float* __restrict__ bias,
                      const float* __restrict__ gumbel,
                      float* __restrict__ logits_out) {
    const int lane = threadIdx.x & 31;
    const int wave = threadIdx.x >> 5;                 // 0..7
    const int l16  = lane & 15;
    const int half = lane >> 4;                        // 0|1
    const int kbase = half * 8;

    const int m0 = blockIdx.y * 64 + (wave >> 2) * 32; // M tile origin (32 rows)
    const int n0 = blockIdx.x * 256 + (wave & 3) * 64; // N tile origin (64 cols)

    v8f acc[2][4];
#pragma unroll
    for (int mi = 0; mi < 2; ++mi)
#pragma unroll
        for (int j = 0; j < 4; ++j) acc[mi][j] = (v8f){};

    const float*  ar0 = A  + (size_t)(m0 +  0 + l16) * DIN + kbase;
    const float*  ar1 = A  + (size_t)(m0 + 16 + l16) * DIN + kbase;
    const __bf16* bp0 = Wt + (size_t)(n0 +  0 + l16) * DIN + kbase;
    const __bf16* bp1 = Wt + (size_t)(n0 + 16 + l16) * DIN + kbase;
    const __bf16* bp2 = Wt + (size_t)(n0 + 32 + l16) * DIN + kbase;
    const __bf16* bp3 = Wt + (size_t)(n0 + 48 + l16) * DIN + kbase;

    // ---- prologue: stage k = 0 ----
    v16bf a0 = load_f32_frag_cvt(ar0);
    v16bf a1 = load_f32_frag_cvt(ar1);
    v16bf b0 = load_bf16_frag(bp0);
    v16bf b1 = load_bf16_frag(bp1);
    v16bf b2 = load_bf16_frag(bp2);
    v16bf b3 = load_bf16_frag(bp3);

    for (int k0 = 32; k0 < DIN; k0 += 32) {
        // prefetch next k-step while current WMMAs execute
        const v16bf na0 = load_f32_frag_cvt(ar0 + k0);
        const v16bf na1 = load_f32_frag_cvt(ar1 + k0);
        const v16bf nb0 = load_bf16_frag(bp0 + k0);
        const v16bf nb1 = load_bf16_frag(bp1 + k0);
        const v16bf nb2 = load_bf16_frag(bp2 + k0);
        const v16bf nb3 = load_bf16_frag(bp3 + k0);

        acc[0][0] = WMMA_BF16(acc[0][0], a0, b0);
        acc[0][1] = WMMA_BF16(acc[0][1], a0, b1);
        acc[0][2] = WMMA_BF16(acc[0][2], a0, b2);
        acc[0][3] = WMMA_BF16(acc[0][3], a0, b3);
        acc[1][0] = WMMA_BF16(acc[1][0], a1, b0);
        acc[1][1] = WMMA_BF16(acc[1][1], a1, b1);
        acc[1][2] = WMMA_BF16(acc[1][2], a1, b2);
        acc[1][3] = WMMA_BF16(acc[1][3], a1, b3);

        a0 = na0; a1 = na1;
        b0 = nb0; b1 = nb1; b2 = nb2; b3 = nb3;
    }
    // ---- epilogue k-step ----
    acc[0][0] = WMMA_BF16(acc[0][0], a0, b0);
    acc[0][1] = WMMA_BF16(acc[0][1], a0, b1);
    acc[0][2] = WMMA_BF16(acc[0][2], a0, b2);
    acc[0][3] = WMMA_BF16(acc[0][3], a0, b3);
    acc[1][0] = WMMA_BF16(acc[1][0], a1, b0);
    acc[1][1] = WMMA_BF16(acc[1][1], a1, b1);
    acc[1][2] = WMMA_BF16(acc[1][2], a1, b2);
    acc[1][3] = WMMA_BF16(acc[1][3], a1, b3);

    // Epilogue: C/D layout — lanes 0-15: M=r, lanes 16-31: M=r+8; N = l16.
#pragma unroll
    for (int mi = 0; mi < 2; ++mi) {
        const int mbase = m0 + mi * 16 + half * 8;
#pragma unroll
        for (int j = 0; j < 4; ++j) {
            const int n = n0 + j * 16 + l16;
            const float bn = bias[n];
#pragma unroll
            for (int r = 0; r < 8; ++r) {
                const size_t idx = (size_t)(mbase + r) * NTOT + n;
                logits_out[idx] = (acc[mi][j][r] + bn + gumbel[idx]) * INV_TAU;
            }
        }
    }
}

// ------------------------------------------------------------------
// Kernel 3: per (row = b*T*G + t*G + g): in-place softmax over V=1024,
// first-occurrence argmax, padding mask, codebook gather.
// Grid = BT*GN blocks, 256 threads; each thread owns 4 consecutive v.
// ------------------------------------------------------------------
__global__ __launch_bounds__(256)
void softmax_argmax_quantize(float* __restrict__ probs,          // logits in, probs out
                             const int* __restrict__ paddings,   // (BT)
                             const float* __restrict__ codebook, // (GN*VN*DN)
                             float* __restrict__ out_ids,        // (BT*GN) as float
                             float* __restrict__ out_q) {        // (BT*GN*DN)
    __shared__ float sval[256];
    __shared__ int   sidx[256];

    const int row = blockIdx.x;     // m*GN + g
    const int m   = row >> 2;       // GN == 4
    const int g   = row & 3;
    const int tid = threadIdx.x;

    float* rp = probs + (size_t)row * VN;
    const float4 x = *(const float4*)(rp + tid * 4);

    // local argmax, first occurrence wins
    float mv = x.x; int mi = tid * 4;
    if (x.y > mv) { mv = x.y; mi = tid * 4 + 1; }
    if (x.z > mv) { mv = x.z; mi = tid * 4 + 2; }
    if (x.w > mv) { mv = x.w; mi = tid * 4 + 3; }

    sval[tid] = mv; sidx[tid] = mi;
    __syncthreads();
#pragma unroll
    for (int s = 128; s > 0; s >>= 1) {
        if (tid < s) {
            const float bv = sval[tid + s]; const int bi = sidx[tid + s];
            const float av = sval[tid];     const int ai = sidx[tid];
            if (bv > av || (bv == av && bi < ai)) { sval[tid] = bv; sidx[tid] = bi; }
        }
        __syncthreads();
    }
    const float rowmax = sval[0];
    const int   rid    = sidx[0];
    __syncthreads();

    // softmax denominator
    const float e0 = __expf(x.x - rowmax);
    const float e1 = __expf(x.y - rowmax);
    const float e2 = __expf(x.z - rowmax);
    const float e3 = __expf(x.w - rowmax);
    sval[tid] = e0 + e1 + e2 + e3;
    __syncthreads();
#pragma unroll
    for (int s = 128; s > 0; s >>= 1) {
        if (tid < s) sval[tid] += sval[tid + s];
        __syncthreads();
    }
    const float denom = sval[0];

    const int   pad   = paddings[m];
    const float scale = pad ? 0.0f : (1.0f / denom);

    float4 p; p.x = e0 * scale; p.y = e1 * scale; p.z = e2 * scale; p.w = e3 * scale;
    *(float4*)(rp + tid * 4) = p;

    if (tid == 0) out_ids[row] = pad ? -1.0f : (float)rid;
    if (tid < DN) {
        out_q[(size_t)row * DN + tid] =
            pad ? 0.0f : codebook[((size_t)g * VN + rid) * DN + tid];
    }
}

// ------------------------------------------------------------------
extern "C" void kernel_launch(void* const* d_in, const int* in_sizes, int n_in,
                              void* d_out, int out_size, void* d_ws, size_t ws_size,
                              hipStream_t stream) {
    const float* inputs   = (const float*)d_in[0];  // (BT, DIN)
    const int*   paddings = (const int*)  d_in[1];  // (BT)
    const float* gumbel   = (const float*)d_in[2];  // (BT, NTOT)
    const float* W        = (const float*)d_in[3];  // (DIN, NTOT)
    const float* bias     = (const float*)d_in[4];  // (NTOT)
    const float* codebook = (const float*)d_in[5];  // (GN, VN, DN)

    float* out       = (float*)d_out;
    float* out_ids   = out;                                            // BT*GN
    float* out_q     = out + (size_t)BT * GN;                          // BT*GN*DN
    float* out_probs = out + (size_t)BT * GN + (size_t)BT * GN * DN;   // BT*GN*VN

    __bf16* Wt = (__bf16*)d_ws;   // NTOT x DIN bf16 = 8 MB

    // 1) W^T in bf16 for contiguous WMMA B-fragment loads
    transpose_convert_w<<<dim3(NTOT / 32, DIN / 32), 256, 0, stream>>>(W, Wt);

    // 2) WMMA GEMM (32x64 wave tiles, SW-pipelined) + fused epilogue
    gemm_logits_wmma<<<dim3(NTOT / 256, BT / 64), 256, 0, stream>>>(
        inputs, Wt, bias, gumbel, out_probs);

    // 3) softmax / argmax / padding mask / codebook gather
    softmax_argmax_quantize<<<dim3(BT * GN), 256, 0, stream>>>(
        out_probs, paddings, codebook, out_ids, out_q);
}
